// TorchRNNDecoder_17042430230591
// MI455X (gfx1250) — matrix operand order repaired and checked by reading, not compiled, not run on hardware
//
#include <hip/hip_runtime.h>
#include <hip/hip_bf16.h>
#include <cstdint>
#include <cstddef>

typedef __attribute__((ext_vector_type(2)))  float    v2f;
typedef __attribute__((ext_vector_type(8)))  float    v8f;
typedef __attribute__((ext_vector_type(8)))  _Float16 v8h;
typedef __attribute__((ext_vector_type(16))) _Float16 v16h;

#define NB    4
#define TIq   512
#define TOP   65            // TO+1
#define VSZ   1024
#define HSZ   512
#define JSZ   512
#define G4H   2048
#define MROW  (NB * TOP)    // 260
#define LDSP  520           // padded f16 row stride for joint A tiles
#define BSWZ_N (512 * 1024) // swizzled out_w halves

// ---------------- grid-wide barrier (persistent co-resident blocks) --------
__device__ __forceinline__ void grid_barrier(unsigned* cnt, unsigned* gen, unsigned nb) {
  __syncthreads();
  if (threadIdx.x == 0) {
    __threadfence();
    unsigned g = __hip_atomic_load(gen, __ATOMIC_RELAXED, __HIP_MEMORY_SCOPE_AGENT);
    unsigned arrived = __hip_atomic_fetch_add(cnt, 1u, __ATOMIC_ACQ_REL, __HIP_MEMORY_SCOPE_AGENT);
    if (arrived == nb - 1u) {
      __hip_atomic_store(cnt, 0u, __ATOMIC_RELAXED, __HIP_MEMORY_SCOPE_AGENT);
      __hip_atomic_store(gen, g + 1u, __ATOMIC_RELEASE, __HIP_MEMORY_SCOPE_AGENT);
    } else {
      while (__hip_atomic_load(gen, __ATOMIC_ACQUIRE, __HIP_MEMORY_SCOPE_AGENT) == g) {
        __builtin_amdgcn_s_sleep(8);
      }
    }
  }
  __syncthreads();
  __threadfence();
}

__device__ __forceinline__ float sigm(float x) { return 1.0f / (1.0f + expf(-x)); }

// ---------------- prep: bias sums, f16-swizzled out_w^T, zero h16, cnt -----
__global__ void prep_kernel(const float* __restrict__ bih0, const float* __restrict__ bhh0,
                            const float* __restrict__ bih1, const float* __restrict__ bhh1,
                            const float* __restrict__ bih2, const float* __restrict__ bhh2,
                            const float* __restrict__ out_w,
                            float* __restrict__ bsum, float* __restrict__ h16,
                            _Float16* __restrict__ bswz, unsigned* __restrict__ cnt) {
  const int idx = blockIdx.x * blockDim.x + threadIdx.x;
  if (idx < BSWZ_N) {
    // bswz[((kt*64 + vt)*32 + lane)*16 + j] = out_w[vt*16 + lane%16][kt*32 + 16*(lane/16) + j]
    const int j    = idx & 15;
    const int lane = (idx >> 4) & 31;
    const int vt   = (idx >> 9) & 63;
    const int kt   = idx >> 15;
    const int K    = kt * 32 + ((lane >> 4) << 4) + j;
    const int Vc   = vt * 16 + (lane & 15);
    bswz[idx] = (_Float16)out_w[(size_t)Vc * JSZ + K];
    return;
  }
  int i2 = idx - BSWZ_N;
  if (i2 < 3 * G4H) {
    const float* bi = (i2 < G4H) ? bih0 : ((i2 < 2 * G4H) ? bih1 : bih2);
    const float* bh = (i2 < G4H) ? bhh0 : ((i2 < 2 * G4H) ? bhh1 : bhh2);
    const int j = i2 & (G4H - 1);
    bsum[i2] = bi[j] + bh[j];
    return;
  }
  int i3 = i2 - 3 * G4H;
  if (i3 < 16 * HSZ) { h16[i3] = 0.0f; return; }
  if (i3 == 16 * HSZ) { *cnt = 0u; }
}

// ---------------- embedding gather (prepend BLANK=0) -----------------------
__global__ void embed_gather(const int* __restrict__ tgt, const float* __restrict__ embed,
                             float* __restrict__ x0) {
  const int idx = blockIdx.x * blockDim.x + threadIdx.x;
  if (idx >= MROW * HSZ) return;
  const int e   = idx & (HSZ - 1);
  const int row = idx >> 9;
  const int n   = row / TOP;
  const int t   = row % TOP;
  const int tok = (t == 0) ? 0 : tgt[n * 64 + (t - 1)];
  x0[idx] = embed[(size_t)tok * HSZ + e];
}

// ---------------- generic fp32 WMMA GEMM: C[M,N] = A[M,K] @ W[N,K]^T (+bias)
// OOB M rows: load address clamped to a valid row (garbage results are
// masked at the store), so the hot loop has NO predication / exec traffic.
__global__ void gemm_f32_wmma(const float* __restrict__ A, const float* __restrict__ W,
                              const float* __restrict__ bias, float* __restrict__ C,
                              int M, int N, int K) {
  const int lane = threadIdx.x & 31;
  const int wave = threadIdx.x >> 5;
  const int lm   = lane & 15;
  const int hi   = lane >> 4;
  const int m0   = blockIdx.x << 4;
  const int n0   = ((blockIdx.y << 3) + wave) << 4;
  const int arow = min(m0 + lm, M - 1);            // clamp instead of predicate
  const float* ap = A + (size_t)arow * K + (hi << 1);
  const float* wp = W + (size_t)(n0 + lm) * K + (hi << 1);
  v8f acc = {0.f, 0.f, 0.f, 0.f, 0.f, 0.f, 0.f, 0.f};

  // software-pipelined K loop
  v2f a0 = *(const v2f*)(ap);
  v2f b0 = *(const v2f*)(wp);
#pragma unroll 8
  for (int k0 = 0; k0 < K - 4; k0 += 4) {
    const v2f a1 = *(const v2f*)(ap + k0 + 4);
    const v2f b1 = *(const v2f*)(wp + k0 + 4);
    acc = __builtin_amdgcn_wmma_f32_16x16x4_f32(false, a0, false, b0, (short)0, acc, false, false);
    a0 = a1; b0 = b1;
  }
  acc = __builtin_amdgcn_wmma_f32_16x16x4_f32(false, a0, false, b0, (short)0, acc, false, false);

  const int col = n0 + lm;
  const float cb = bias ? bias[col] : 0.0f;
#pragma unroll
  for (int r = 0; r < 8; ++r) {
    const int m = m0 + r + (hi << 3);
    if (m < M) C[(size_t)m * N + col] = acc[r] + cb;
  }
}

// ---------------- persistent LSTM layer: 4 blocks x 1024 threads -----------
// xg[(n*65+t)*2048] holds x@Wih^T + bih + bhh (precomputed, all t).
// Serial loop over t: phase A  g = h @ Whh^T  (fp32 WMMA, batch padded 4->16),
// phase B gate math (block-local via LDS), one grid barrier per step.
__global__ void __launch_bounds__(1024, 1)
lstm_layer(const float* __restrict__ xg, const float* __restrict__ Whh,
           float* __restrict__ h16, float* __restrict__ xout,
           unsigned* __restrict__ cnt, unsigned* __restrict__ gen) {
  __shared__ float glds[4 * 512];   // gate preacts for this block's 128 h-cols
  __shared__ float clds[512];       // cell state  (n,lh)
  const int tid  = threadIdx.x;
  const int lane = tid & 31;
  const int wave = tid >> 5;        // 0..31
  const int lm   = lane & 15;
  const int hi   = lane >> 4;
  const int htL  = wave & 7;        // local h-tile 0..7
  const int gate = wave >> 3;       // 0..3 (i,f,g,o)
  const int hc0  = (blockIdx.x * 8 + htL) * 16;   // global h col base
  const int j0   = gate * HSZ + hc0;              // gate-matrix column base
  const float* hp = h16 + lm * HSZ + (hi << 1);   // rows 4..15 pre-zeroed
  const float* wp = Whh + (size_t)(j0 + lm) * HSZ + (hi << 1);

  for (int t = 0; t < TOP; ++t) {
    if (t > 0) {
      v8f acc = {0.f, 0.f, 0.f, 0.f, 0.f, 0.f, 0.f, 0.f};
      v2f a0 = *(const v2f*)(hp);
      v2f b0 = *(const v2f*)(wp);
#pragma unroll 8
      for (int k0 = 0; k0 < HSZ - 4; k0 += 4) {
        const v2f a1 = *(const v2f*)(hp + k0 + 4);
        const v2f b1 = *(const v2f*)(wp + k0 + 4);
        acc = __builtin_amdgcn_wmma_f32_16x16x4_f32(false, a0, false, b0, (short)0, acc, false, false);
        a0 = a1; b0 = b1;
      }
      acc = __builtin_amdgcn_wmma_f32_16x16x4_f32(false, a0, false, b0, (short)0, acc, false, false);
      if (hi == 0) {                 // rows m=0..3 live in acc[0..3], lanes 0-15
        const int cc = gate * 128 + htL * 16 + lm;
        glds[0 * 512 + cc] = acc[0];
        glds[1 * 512 + cc] = acc[1];
        glds[2 * 512 + cc] = acc[2];
        glds[3 * 512 + cc] = acc[3];
      }
    }
    __syncthreads();
    if (tid < 512) {
      const int lh = tid & 127;
      const int n  = tid >> 7;
      const int h  = blockIdx.x * 128 + lh;
      const size_t xr = (size_t)(n * TOP + t) * G4H;
      float gi = xg[xr + h];
      float gf = xg[xr + HSZ + h];
      float gc = xg[xr + 2 * HSZ + h];
      float go = xg[xr + 3 * HSZ + h];
      if (t > 0) {
        gi += glds[n * 512 + lh];
        gf += glds[n * 512 + 128 + lh];
        gc += glds[n * 512 + 256 + lh];
        go += glds[n * 512 + 384 + lh];
      }
      const float i_ = sigm(gi), f_ = sigm(gf), o_ = sigm(go);
      const float g_ = tanhf(gc);
      const float cp = (t > 0) ? clds[tid] : 0.0f;
      const float cn = f_ * cp + i_ * g_;
      const float hn = o_ * tanhf(cn);
      clds[tid] = cn;
      h16[n * HSZ + h] = hn;
      xout[(size_t)(n * TOP + t) * HSZ + h] = hn;
    }
    grid_barrier(cnt, gen, gridDim.x);   // publish h before next step's matmul
  }
}

// ---------------- joint: out[n,t,u,v] = tanh(encp+decp) @ out_w^T ----------
// Block: 16 t-rows x 2 u-values x all 1024 v.  A tiles (f16) staged in LDS,
// B (out_w^T) pre-swizzled f16 frags; f16 WMMA 16x16x32 with f32 accumulate.
__global__ void __launch_bounds__(256)
joint_wmma(const float* __restrict__ encp, const float* __restrict__ decp,
           const _Float16* __restrict__ bswz, float* __restrict__ out) {
  __shared__ _Float16 a0s[16 * LDSP];
  __shared__ _Float16 a1s[16 * LDSP];
  const int t0 = blockIdx.x * 16;
  const int u0 = blockIdx.y * 2;
  const int u1 = u0 + 1;
  const int n  = blockIdx.z;
  const int tid = threadIdx.x;

  const float* ep = encp + (size_t)(n * TIq + t0) * JSZ;
  const float* d0 = decp + (size_t)(n * TOP + u0) * JSZ;
  const float* d1 = (u1 < TOP) ? (decp + (size_t)(n * TOP + u1) * JSZ) : d0;
  for (int idx = tid; idx < 16 * JSZ; idx += 256) {
    const int m = idx >> 9;
    const int j = idx & (JSZ - 1);
    const float e = ep[m * JSZ + j];
    a0s[m * LDSP + j] = (_Float16)tanhf(e + d0[j]);
    a1s[m * LDSP + j] = (_Float16)tanhf(e + d1[j]);
  }
  __syncthreads();

  const int lane = tid & 31;
  const int wave = tid >> 5;
  const int lm   = lane & 15;
  const int hic  = lane >> 4;
  for (int vt = 0; vt < 8; ++vt) {
    const int v0 = ((wave << 3) + vt) << 4;
    v8f acc0 = {0.f, 0.f, 0.f, 0.f, 0.f, 0.f, 0.f, 0.f};
    v8f acc1 = {0.f, 0.f, 0.f, 0.f, 0.f, 0.f, 0.f, 0.f};
#pragma unroll
    for (int kt = 0; kt < 16; ++kt) {
      const v16h b = *(const v16h*)(bswz + ((((size_t)kt * 64 + (v0 >> 4)) * 32 + lane) << 4));
      union { v16h v; v8h h[2]; } a0u, a1u;
      const int off = lm * LDSP + kt * 32 + (hic << 4);
      a0u.h[0] = *(const v8h*)(a0s + off);
      a0u.h[1] = *(const v8h*)(a0s + off + 8);
      a1u.h[0] = *(const v8h*)(a1s + off);
      a1u.h[1] = *(const v8h*)(a1s + off + 8);
      acc0 = __builtin_amdgcn_wmma_f32_16x16x32_f16(false, a0u.v, false, b, (short)0, acc0, false, false);
      acc1 = __builtin_amdgcn_wmma_f32_16x16x32_f16(false, a1u.v, false, b, (short)0, acc1, false, false);
    }
    const int vcol = v0 + lm;
#pragma unroll
    for (int r = 0; r < 8; ++r) {
      const int t = t0 + r + (hic << 3);
      const size_t base = ((size_t)(n * TIq + t)) * TOP;
      out[(base + u0) * VSZ + vcol] = acc0[r];
      if (u1 < TOP) out[(base + u1) * VSZ + vcol] = acc1[r];
    }
  }
}

// ---------------------------------------------------------------------------
extern "C" void kernel_launch(void* const* d_in, const int* in_sizes, int n_in,
                              void* d_out, int out_size, void* d_ws, size_t ws_size,
                              hipStream_t stream) {
  (void)in_sizes; (void)n_in; (void)out_size; (void)ws_size;
  const float* enc_out = (const float*)d_in[0];
  const int*   tgt     = (const int*)d_in[1];
  const float* embed   = (const float*)d_in[2];
  const float* Wih[3]  = {(const float*)d_in[3], (const float*)d_in[7],  (const float*)d_in[11]};
  const float* Whh[3]  = {(const float*)d_in[4], (const float*)d_in[8],  (const float*)d_in[12]};
  const float* bih[3]  = {(const float*)d_in[5], (const float*)d_in[9],  (const float*)d_in[13]};
  const float* bhh[3]  = {(const float*)d_in[6], (const float*)d_in[10], (const float*)d_in[14]};
  const float* enc_w   = (const float*)d_in[15];
  const float* dec_w   = (const float*)d_in[16];
  const float* dec_b   = (const float*)d_in[17];
  const float* out_w   = (const float*)d_in[18];
  float* out = (float*)d_out;

  // workspace carving (256B aligned)
  char* w = (char*)d_ws;
  auto carve = [&](size_t bytes) -> void* {
    void* p = (void*)w;
    w += (bytes + 255) & ~(size_t)255;
    return p;
  };
  float*     xbufA = (float*)carve((size_t)MROW * HSZ * 4);       // 260x512
  float*     xbufB = (float*)carve((size_t)MROW * HSZ * 4);
  float*     xg    = (float*)carve((size_t)MROW * G4H * 4);       // 260x2048
  float*     bsum  = (float*)carve((size_t)3 * G4H * 4);
  float*     h16   = (float*)carve((size_t)16 * HSZ * 4);         // padded h
  float*     encp  = (float*)carve((size_t)NB * TIq * JSZ * 4);   // 4MB
  float*     decp  = (float*)carve((size_t)MROW * JSZ * 4);
  _Float16*  bswz  = (_Float16*)carve((size_t)BSWZ_N * 2);        // 1MB
  unsigned*  sync  = (unsigned*)carve(256);
  unsigned*  cnt   = sync;
  unsigned*  gen   = sync + 1;

  // 1) prep: bias sums, swizzled f16 out_w^T, zero padded-h, reset barrier cnt
  {
    const int total = BSWZ_N + 3 * G4H + 16 * HSZ + 1;
    prep_kernel<<<(total + 255) / 256, 256, 0, stream>>>(
        bih[0], bhh[0], bih[1], bhh[1], bih[2], bhh[2], out_w, bsum, h16, bswz, cnt);
  }
  // 2) embedding gather (prepend blank)
  embed_gather<<<(MROW * HSZ + 255) / 256, 256, 0, stream>>>(tgt, embed, xbufA);

  // 3) 3 LSTM layers: parallel input GEMM, then persistent serial recurrence
  float* xin  = xbufA;
  float* xou  = xbufB;
  for (int l = 0; l < 3; ++l) {
    gemm_f32_wmma<<<dim3(17, 16), 256, 0, stream>>>(xin, Wih[l], bsum + l * G4H, xg,
                                                    MROW, G4H, HSZ);
    lstm_layer<<<4, 1024, 0, stream>>>(xg, Whh[l], h16, xou, cnt, gen);
    float* tmp = xin; xin = xou; xou = tmp;
  }
  // xin now holds the final layer's hidden states [260 x 512]

  // 4) projections
  gemm_f32_wmma<<<dim3(128, 4), 256, 0, stream>>>(enc_out, enc_w, nullptr, encp,
                                                  NB * TIq, JSZ, HSZ);
  gemm_f32_wmma<<<dim3(17, 4), 256, 0, stream>>>(xin, dec_w, dec_b, decp,
                                                 MROW, JSZ, HSZ);
  // 5) joint network (f16 WMMA, f32 accumulate)
  joint_wmma<<<dim3(TIq / 16, (TOP + 1) / 2, NB), 256, 0, stream>>>(encp, decp, bswz, out);
}
